// GATLayer_79259326480595
// MI455X (gfx1250) — matrix operand order repaired and checked by reading, not compiled, not run on hardware
//
#include <hip/hip_runtime.h>
#include <hip/hip_bf16.h>

#define NB 8
#define NN 1024
#define FIN 256
#define FOUT 64
#define NEGV (-9.0e15f)

typedef __attribute__((ext_vector_type(16))) _Float16 v16h;
typedef __attribute__((ext_vector_type(8)))  float    v8f;

// ---------------------------------------------------------------------------
// WMMA fragment helpers (layouts per CDNA5 ISA 7.12.2, wave32)
// A 16x32 f16: lanes 0-15 hold M=lane, K {0..7,16..23}; lanes 16-31 hold
// M=lane-16, K {8..15,24..31}.
__device__ __forceinline__ v16h load_a_frag(const _Float16* lds, int stride,
                                            int row0, int k0, int lane) {
  const int m  = row0 + (lane & 15);
  const int kh = (lane >> 4) * 8;
  const _Float16* p = lds + m * stride + k0;
  v16h a;
#pragma unroll
  for (int i = 0; i < 8; ++i) a[i] = p[kh + i];
#pragma unroll
  for (int i = 0; i < 8; ++i) a[i + 8] = p[16 + kh + i];
  return a;
}

// B 32x16 f16 from column-major LDS tile Bs[col][k]: lane = col (mod 16),
// lane-half h holds K {8h..8h+7, 16+8h..16+8h+7} (same order as A).
__device__ __forceinline__ v16h load_b_frag(const _Float16* lds, int stride,
                                            int col0, int k0, int lane) {
  const int c  = col0 + (lane & 15);
  const int kb = k0 + (lane >> 4) * 8;
  const _Float16* p = lds + c * stride + kb;
  v16h b;
#pragma unroll
  for (int i = 0; i < 16; ++i) b[i] = p[i < 8 ? i : i + 8];
  return b;
}

__device__ __forceinline__ v8f wmma_f16(v16h a, v16h b, v8f c) {
  return __builtin_amdgcn_wmma_f32_16x16x32_f16(false, a, false, b,
                                                (short)0, c, false, false);
}

// ---------------------------------------------------------------------------
// K0: h = x@W (f32), plus f16 copy, sq=|h|^2, Wh1=h.a1, Wh2=h.a2
__global__ __launch_bounds__(64) void h_kernel(
    const float* __restrict__ x, const float* __restrict__ W,
    const float* __restrict__ a, _Float16* __restrict__ hH,
    float* __restrict__ sq, float* __restrict__ Wh1, float* __restrict__ Wh2) {
  const int row = blockIdx.x;          // b*N + n
  const int o = threadIdx.x;           // 0..63
  __shared__ float xs[FIN];
  for (int k = o; k < FIN; k += 64) xs[k] = x[(size_t)row * FIN + k];
  __syncthreads();
  float acc = 0.f;
#pragma unroll 8
  for (int k = 0; k < FIN; ++k) acc = fmaf(xs[k], W[k * FOUT + o], acc);
  hH[(size_t)row * FOUT + o] = (_Float16)acc;

  __shared__ float r0[64], r1[64], r2[64];
  r0[o] = acc * acc; r1[o] = acc * a[o]; r2[o] = acc * a[FOUT + o];
  __syncthreads();
  for (int s = 32; s > 0; s >>= 1) {
    if (o < s) { r0[o] += r0[o + s]; r1[o] += r1[o + s]; r2[o] += r2[o + s]; }
    __syncthreads();
  }
  if (o == 0) { sq[row] = r0[0]; Wh1[row] = r1[0]; Wh2[row] = r2[0]; }
}

// ---------------------------------------------------------------------------
// K1: adj_raw = exp(-dist) with G = h.h^T via WMMA (K=64). 64x64 tile/block.
__global__ __launch_bounds__(256) void pairwise_adj_kernel(
    const _Float16* __restrict__ hH, const float* __restrict__ sq,
    float* __restrict__ adj_raw) {
  const int b = blockIdx.z;
  const int row0 = blockIdx.y * 64, col0 = blockIdx.x * 64;
  const int tid = threadIdx.x, lane = tid & 31, wave = tid >> 5;
  const size_t bOff = (size_t)b * NN * NN;

  __shared__ __align__(16) _Float16 As[64][72];
  __shared__ __align__(16) _Float16 Bs[64][72];

  {  // 16 f16 per thread (two b128 each), full K=64 tiles
    const int r = tid >> 2, kk = (tid & 3) * 16;
    const _Float16* sa = hH + (size_t)(b * NN + row0 + r) * FOUT + kk;
    const _Float16* sb = hH + (size_t)(b * NN + col0 + r) * FOUT + kk;
    ((uint4*)&As[r][kk])[0] = ((const uint4*)sa)[0];
    ((uint4*)&As[r][kk])[1] = ((const uint4*)sa)[1];
    ((uint4*)&Bs[r][kk])[0] = ((const uint4*)sb)[0];
    ((uint4*)&Bs[r][kk])[1] = ((const uint4*)sb)[1];
  }
  __syncthreads();

  const int m0 = (wave >> 2) * 16;      // tiles wave and wave+8 share column
  const int c0 = (wave & 3) * 16;
  v8f acc0 = {}, acc1 = {};
#pragma unroll
  for (int k0 = 0; k0 < 64; k0 += 32) {
    v16h bf = load_b_frag(&Bs[0][0], 72, c0, k0, lane);
    v16h a0 = load_a_frag(&As[0][0], 72, m0, k0, lane);
    v16h a1 = load_a_frag(&As[0][0], 72, m0 + 32, k0, lane);
    acc0 = wmma_f16(a0, bf, acc0);
    acc1 = wmma_f16(a1, bf, acc1);
  }

  const int mr = (lane >> 4) * 8, nc = lane & 15;
  const int j = col0 + c0 + nc;
  const float sqj = sq[b * NN + j];
#pragma unroll
  for (int r = 0; r < 8; ++r) {
    const int i0 = row0 + m0 + mr + r;
    const int i1 = i0 + 32;
    float d2a = sq[b * NN + i0] + sqj - 2.f * acc0[r];
    float d2b = sq[b * NN + i1] + sqj - 2.f * acc1[r];
    float da = d2a > 0.f ? sqrtf(d2a) : 0.f;
    float db = d2b > 0.f ? sqrtf(d2b) : 0.f;
    adj_raw[bOff + (size_t)i0 * NN + j] = expf(-da);
    adj_raw[bOff + (size_t)i1 * NN + j] = expf(-db);
  }
}

// ---------------------------------------------------------------------------
// K2: row-normalize in place + f16 copy. One block per (b,i) row.
__global__ __launch_bounds__(256) void rownorm_kernel(
    float* __restrict__ adj, _Float16* __restrict__ adjh) {
  const size_t base = (size_t)blockIdx.x * NN;
  const int tid = threadIdx.x;
  float v[4]; float s = 0.f;
#pragma unroll
  for (int q = 0; q < 4; ++q) { v[q] = adj[base + tid + q * 256]; s += v[q]; }
  __shared__ float red[256];
  red[tid] = s; __syncthreads();
  for (int st = 128; st > 0; st >>= 1) {
    if (tid < st) red[tid] += red[tid + st];
    __syncthreads();
  }
  const float scale = 1.f / fmaxf(red[0], 1e-12f);
#pragma unroll
  for (int q = 0; q < 4; ++q) {
    const float nv = v[q] * scale;
    adj[base + tid + q * 256] = nv;
    adjh[base + tid + q * 256] = (_Float16)nv;
  }
}

// ---------------------------------------------------------------------------
// K3: tiled f16 WMMA GEMM C = A*B (1024x1024x1024 per batch), 64x64/block.
// mode 0: Cf = C, Ch = (f16)C          (adj2)
// mode 1: Cf = Cf + 0.8*Adj2 + 0.64*C  (S, in place over adj)
__global__ __launch_bounds__(256) void gemm_adj_kernel(
    const _Float16* __restrict__ A, const _Float16* __restrict__ Bm,
    float* __restrict__ Cf, _Float16* __restrict__ Ch,
    const float* __restrict__ Adj2, int mode) {
  const int b = blockIdx.z;
  const int row0 = blockIdx.y * 64, col0 = blockIdx.x * 64;
  const int tid = threadIdx.x, lane = tid & 31, wave = tid >> 5;
  const size_t bOff = (size_t)b * NN * NN;

  __shared__ __align__(16) _Float16 As[64][40];   // [m][k]
  __shared__ __align__(16) _Float16 Bs[64][40];   // [n][k] (transposed stage)

  const int ar = tid >> 2, ac = (tid & 3) * 8;    // A: 8 f16 per thread
  const int bk = tid >> 3, bc = (tid & 7) * 8;    // B: 8 f16 per thread
  const int m0 = (wave >> 2) * 16;
  const int c0 = (wave & 3) * 16;
  v8f acc0 = {}, acc1 = {};

  for (int k0 = 0; k0 < NN; k0 += 32) {
    __syncthreads();
    *(uint4*)&As[ar][ac] =
        *(const uint4*)&A[bOff + (size_t)(row0 + ar) * NN + k0 + ac];
    {
      _Float16 tmp[8];
      *(uint4*)tmp =
          *(const uint4*)&Bm[bOff + (size_t)(k0 + bk) * NN + col0 + bc];
#pragma unroll
      for (int i = 0; i < 8; ++i) Bs[bc + i][bk] = tmp[i];
    }
    if (k0 + 32 < NN) {  // pull next A tile toward L2/L0 early
      __builtin_prefetch(&A[bOff + (size_t)(row0 + ar) * NN + k0 + 32 + ac], 0, 0);
    }
    __syncthreads();

    v16h bf = load_b_frag(&Bs[0][0], 40, c0, 0, lane);
    v16h a0 = load_a_frag(&As[0][0], 40, m0, 0, lane);
    v16h a1 = load_a_frag(&As[0][0], 40, m0 + 32, 0, lane);
    acc0 = wmma_f16(a0, bf, acc0);
    acc1 = wmma_f16(a1, bf, acc1);
  }

  const int mr = (lane >> 4) * 8, nc = lane & 15;
  const int j = col0 + c0 + nc;
#pragma unroll
  for (int r = 0; r < 8; ++r) {
    const int i0 = row0 + m0 + mr + r;
    const int i1 = i0 + 32;
    const size_t x0 = bOff + (size_t)i0 * NN + j;
    const size_t x1 = bOff + (size_t)i1 * NN + j;
    if (mode == 0) {
      Cf[x0] = acc0[r]; Ch[x0] = (_Float16)acc0[r];
      Cf[x1] = acc1[r]; Ch[x1] = (_Float16)acc1[r];
    } else {
      Cf[x0] = Cf[x0] + 0.8f * Adj2[x0] + 0.64f * acc0[r];
      Cf[x1] = Cf[x1] + 0.8f * Adj2[x1] + 0.64f * acc1[r];
    }
  }
}

// ---------------------------------------------------------------------------
// K4: Ssym = 0.5*(S + S^T)
__global__ __launch_bounds__(256) void symmetrize_kernel(
    const float* __restrict__ S, float* __restrict__ Ssym) {
  const int b = blockIdx.z;
  const int i0 = blockIdx.y * 32, j0 = blockIdx.x * 32;
  const size_t bOff = (size_t)b * NN * NN;
  for (int e = threadIdx.x; e < 1024; e += 256) {
    const int i = i0 + (e >> 5), j = j0 + (e & 31);
    Ssym[bOff + (size_t)i * NN + j] =
        0.5f * (S[bOff + (size_t)i * NN + j] + S[bOff + (size_t)j * NN + i]);
  }
}

// ---------------------------------------------------------------------------
// K5: per-batch threshold = min(diag) - mean(diag[n]-offdiag[n])
__global__ __launch_bounds__(256) void thr_kernel(
    const float* __restrict__ Ssym, float* __restrict__ thr) {
  const int b = blockIdx.x, tid = threadIdx.x;
  const size_t bOff = (size_t)b * NN * NN;
  float mn = 3.4e38f, sum = 0.f;
  for (int n = tid; n < NN; n += 256) {
    const float d = Ssym[bOff + (size_t)n * NN + n];
    mn = fminf(mn, d);
    if (n < NN - 1) sum += d - Ssym[bOff + (size_t)n * NN + n + 1];
  }
  __shared__ float rmn[256], rs[256];
  rmn[tid] = mn; rs[tid] = sum; __syncthreads();
  for (int st = 128; st > 0; st >>= 1) {
    if (tid < st) { rmn[tid] = fminf(rmn[tid], rmn[tid + st]); rs[tid] += rs[tid + st]; }
    __syncthreads();
  }
  if (tid == 0) thr[b] = rmn[0] - rs[0] / (float)(NN - 1);
}

__device__ __forceinline__ float lrelu(float x) { return x > 0.f ? x : 0.01f * x; }

// ---------------------------------------------------------------------------
// K6: batchnorm stats per row-index n over (b, j)
__global__ __launch_bounds__(256) void stats_kernel(
    const float* __restrict__ Ssym, const float* __restrict__ thr,
    const float* __restrict__ Wh1, const float* __restrict__ Wh2,
    float* __restrict__ mean, float* __restrict__ invstd) {
  const int n = blockIdx.x, tid = threadIdx.x;
  float s = 0.f, ss = 0.f;
  for (int b = 0; b < NB; ++b) {
    const float t = thr[b];
    const float w1 = Wh1[b * NN + n];
    const float* row = Ssym + ((size_t)b * NN + n) * NN;
    for (int j = tid; j < NN; j += 256) {
      const float e = lrelu(w1 + Wh2[b * NN + j]);
      const float att = row[j] > t ? e : NEGV;
      s += att; ss += att * att;
    }
  }
  __shared__ float r0[256], r1[256];
  r0[tid] = s; r1[tid] = ss; __syncthreads();
  for (int st = 128; st > 0; st >>= 1) {
    if (tid < st) { r0[tid] += r0[tid + st]; r1[tid] += r1[tid + st]; }
    __syncthreads();
  }
  if (tid == 0) {
    const float m = r0[0] / (float)(NB * NN);
    const float var = r1[0] / (float)(NB * NN) - m * m;
    mean[n] = m;
    invstd[n] = rsqrtf(var + 1e-5f);
  }
}

// ---------------------------------------------------------------------------
// K7: fused mask + BN + row softmax -> out
__global__ __launch_bounds__(256) void softmax_kernel(
    const float* __restrict__ Ssym, const float* __restrict__ thr,
    const float* __restrict__ Wh1, const float* __restrict__ Wh2,
    const float* __restrict__ mean, const float* __restrict__ invstd,
    float* __restrict__ out) {
  const int row = blockIdx.x;               // b*N + n
  const int b = row >> 10, n = row & 1023;
  const int tid = threadIdx.x;
  const float t = thr[b], w1 = Wh1[row], m = mean[n], is = invstd[n];
  const float* srow = Ssym + (size_t)row * NN;
  float y[4]; float mx = -3.4e38f;
#pragma unroll
  for (int q = 0; q < 4; ++q) {
    const int j = tid + q * 256;
    const float e = lrelu(w1 + Wh2[b * NN + j]);
    const float att = srow[j] > t ? e : NEGV;
    y[q] = (att - m) * is;
    mx = fmaxf(mx, y[q]);
  }
  __shared__ float red[256];
  red[tid] = mx; __syncthreads();
  for (int st = 128; st > 0; st >>= 1) {
    if (tid < st) red[tid] = fmaxf(red[tid], red[tid + st]);
    __syncthreads();
  }
  const float rowmax = red[0];
  __syncthreads();
  float s = 0.f;
#pragma unroll
  for (int q = 0; q < 4; ++q) { y[q] = expf(y[q] - rowmax); s += y[q]; }
  red[tid] = s; __syncthreads();
  for (int st = 128; st > 0; st >>= 1) {
    if (tid < st) red[tid] += red[tid + st];
    __syncthreads();
  }
  const float inv = 1.f / red[0];
#pragma unroll
  for (int q = 0; q < 4; ++q) out[(size_t)row * NN + tid + q * 256] = y[q] * inv;
}

// ---------------------------------------------------------------------------
extern "C" void kernel_launch(void* const* d_in, const int* in_sizes, int n_in,
                              void* d_out, int out_size, void* d_ws, size_t ws_size,
                              hipStream_t stream) {
  const float* x = (const float*)d_in[0];   // (8,1024,256)
  const float* W = (const float*)d_in[1];   // (256,64)
  const float* a = (const float*)d_in[2];   // (128,1)
  float* out = (float*)d_out;               // (8,1024,1024)

  // workspace carve-up (~97 MiB total)
  char* w = (char*)d_ws;
  _Float16* hH   = (_Float16*)w; w += (size_t)NB * NN * FOUT * 2;   // 1 MiB
  float* sq      = (float*)w;    w += (size_t)NB * NN * 4;
  float* Wh1     = (float*)w;    w += (size_t)NB * NN * 4;
  float* Wh2     = (float*)w;    w += (size_t)NB * NN * 4;
  float* thr     = (float*)w;    w += 256;
  float* mean    = (float*)w;    w += (size_t)NN * 4;
  float* invstd  = (float*)w;    w += (size_t)NN * 4;
  float* adjPf   = (float*)w;    w += (size_t)NB * NN * NN * 4;     // 32 MiB
  _Float16* adjPh= (_Float16*)w; w += (size_t)NB * NN * NN * 2;     // 16 MiB
  float* adj2f   = (float*)w;    w += (size_t)NB * NN * NN * 4;     // 32 MiB
  _Float16* adj2h= (_Float16*)w; w += (size_t)NB * NN * NN * 2;     // 16 MiB
  (void)ws_size; (void)in_sizes; (void)n_in; (void)out_size;

  // 1. h = x@W, sq, Wh1, Wh2
  h_kernel<<<NB * NN, 64, 0, stream>>>(x, W, a, hH, sq, Wh1, Wh2);
  // 2. adj_raw = exp(-pairwise_dist) via WMMA h.h^T
  pairwise_adj_kernel<<<dim3(NN / 64, NN / 64, NB), 256, 0, stream>>>(hH, sq, adjPf);
  // 3. row-normalize (in place) + f16 copy
  rownorm_kernel<<<NB * NN, 256, 0, stream>>>(adjPf, adjPh);
  // 4. adj2 = adj@adj (WMMA)
  gemm_adj_kernel<<<dim3(NN / 64, NN / 64, NB), 256, 0, stream>>>(
      adjPh, adjPh, adj2f, adj2h, nullptr, 0);
  // 5. S = adj + 0.8*adj2 + 0.64*(adj@adj2)  (WMMA, in place over adjPf)
  gemm_adj_kernel<<<dim3(NN / 64, NN / 64, NB), 256, 0, stream>>>(
      adjPh, adj2h, adjPf, nullptr, adj2f, 1);
  // 6. Ssym = 0.5*(S + S^T)  (adj2f reused as Ssym)
  symmetrize_kernel<<<dim3(NN / 32, NN / 32, NB), 256, 0, stream>>>(adjPf, adj2f);
  // 7. per-batch threshold
  thr_kernel<<<NB, 256, 0, stream>>>(adj2f, thr);
  // 8. BN stats per row index
  stats_kernel<<<NN, 256, 0, stream>>>(adj2f, thr, Wh1, Wh2, mean, invstd);
  // 9. fused mask + BN + softmax
  softmax_kernel<<<NB * NN, 256, 0, stream>>>(adj2f, thr, Wh1, Wh2, mean, invstd, out);
}
